// HOIModel_76364518523366
// MI455X (gfx1250) — compile-verified, matrix operand order checked by reading, not run on hardware
//
#include <hip/hip_runtime.h>
#include <hip/hip_bf16.h>

// ---------------------------------------------------------------------------
// Problem constants (from the reference)
// ---------------------------------------------------------------------------
#define B_    16
#define H_    74
#define W_    74
#define D_    768
#define NH_   8
#define NO_   16
#define NBOX  (NH_ + NO_)        // 24 boxes per batch
#define M_    (B_ * NH_ * NO_)   // 2048 pair rows
#define K1_   (2 * D_)           // 1536
#define N1_   512
#define N2_   256
#define N3_   117
#define N3P_  128                // padded to a 16 multiple for WMMA tiles

// Register-blocking factors: each wave computes a (16*MR) x (16*NR) tile.
#define MR_   4                  // 64 rows
#define NR_   2                  // 32 cols

typedef __attribute__((ext_vector_type(16))) _Float16 v16h;
typedef __attribute__((ext_vector_type(8)))  _Float16 v8h;
typedef __attribute__((ext_vector_type(8)))  float    v8f;

union HFrag { v16h v; v8h h[2]; };

// ---------------------------------------------------------------------------
// 1) ROI mean: one block per (batch, box). Direct region sum (equivalent to
//    the reference's integral-image difference), coalesced over D.
// ---------------------------------------------------------------------------
__global__ void roi_mean_kernel(const float* __restrict__ feats,
                                const int* __restrict__ hboxes,
                                const int* __restrict__ oboxes,
                                float* __restrict__ roi) {
    const int blk = blockIdx.x;          // b*NBOX + box
    const int b   = blk / NBOX;
    const int box = blk - b * NBOX;
    const int* bx = (box < NH_) ? (hboxes + (b * NH_ + box) * 4)
                                : (oboxes + (b * NO_ + (box - NH_)) * 4);
    const int x1 = bx[0], y1 = bx[1], x2 = bx[2], y2 = bx[3];
    const float inv_area = 1.0f / (float)((y2 - y1) * (x2 - x1));

    for (int d = threadIdx.x; d < D_; d += blockDim.x) {
        float s = 0.0f;
        for (int y = y1; y < y2; ++y) {
            const float* row = feats + ((size_t)(b * H_ + y) * W_) * D_;
            for (int x = x1; x < x2; ++x)
                s += row[(size_t)x * D_ + d];
        }
        roi[(size_t)blk * D_ + d] = s * inv_area;
    }
}

// ---------------------------------------------------------------------------
// 2) Build pair matrix A1 (M_ x K1_) in f16: [hf | of] per (b,h,o) row.
// ---------------------------------------------------------------------------
__global__ void build_pairs_kernel(const float* __restrict__ roi,
                                   _Float16* __restrict__ A1) {
    const int p = blockIdx.x;            // 0..M_-1
    const int b = p >> 7;                // / (NH_*NO_)
    const int h = (p >> 4) & (NH_ - 1);
    const int o = p & (NO_ - 1);
    const float* hf = roi + (size_t)(b * NBOX + h) * D_;
    const float* of = roi + (size_t)(b * NBOX + NH_ + o) * D_;
    _Float16* dst = A1 + (size_t)p * K1_;
    for (int k = threadIdx.x; k < D_; k += blockDim.x) {
        dst[k]      = (_Float16)hf[k];
        dst[D_ + k] = (_Float16)of[k];
    }
}

// ---------------------------------------------------------------------------
// 3) Weight transpose + f16 convert: Wt[n*K + k] = W[k*N + n]; rows n>=N zero.
//    N-major storage makes the WMMA B-fragment two contiguous 16B loads.
// ---------------------------------------------------------------------------
__global__ void transpose_w_kernel(const float* __restrict__ W,
                                   _Float16* __restrict__ Wt,
                                   int K, int N, int NP) {
    const int idx = blockIdx.x * blockDim.x + threadIdx.x;
    if (idx >= NP * K) return;
    const int n = idx / K;
    const int k = idx - n * K;
    const float v = (n < N) ? W[(size_t)k * N + n] : 0.0f;
    Wt[idx] = (_Float16)v;
}

// ---------------------------------------------------------------------------
// 4) WMMA GEMM, register-blocked: one wave per 64x32 output tile
//    (MR_ x NR_ grid of 16x16 WMMA tiles -> 8 accumulators, 8 WMMAs per
//    12 b128 loads each K-step, ~2.7x the compute density of 1 tile/wave).
//
//    A: M x K row-major f16.  Bt: N x K row-major f16 (B transposed).
//    Fragment addressing per the CDNA5 ISA 16-bit 16x32 A layout:
//      lanes 0-15 : row M=lane,    halves 0..7 = K 0..7,  8..15 = K 16..23
//      lanes 16-31: row M=lane-16, halves 0..7 = K 8..15, 8..15 = K 24..31
//    C/D f32 layout: VGPR r -> (M = r + 8*laneHi, N = lane&15).
// ---------------------------------------------------------------------------
__global__ void __launch_bounds__(32)
wmma_gemm_kernel(const _Float16* __restrict__ A,
                 const _Float16* __restrict__ Bt,
                 const float* __restrict__ bias,
                 void* __restrict__ out,
                 int K, int nvalid, int ldout, int final_f32) {
    const int t    = threadIdx.x;
    const int lane = t & 15;
    const int hi   = (t >> 4) & 1;

    const int m0 = blockIdx.y * (16 * MR_);
    const int n0 = blockIdx.x * (16 * NR_);

    const _Float16* aBase[MR_];
    const _Float16* bBase[NR_];
    #pragma unroll
    for (int i = 0; i < MR_; ++i)
        aBase[i] = A + (size_t)(m0 + i * 16 + lane) * K + hi * 8;
    #pragma unroll
    for (int j = 0; j < NR_; ++j)
        bBase[j] = Bt + (size_t)(n0 + j * 16 + lane) * K + hi * 8;

    v8f acc[MR_][NR_];
    #pragma unroll
    for (int i = 0; i < MR_; ++i)
        #pragma unroll
        for (int j = 0; j < NR_; ++j)
            acc[i][j] = (v8f){};

    for (int k0 = 0; k0 < K; k0 += 32) {
        HFrag a[MR_], b[NR_];
        #pragma unroll
        for (int i = 0; i < MR_; ++i) {
            a[i].h[0] = *(const v8h*)(aBase[i] + k0);
            a[i].h[1] = *(const v8h*)(aBase[i] + k0 + 16);
        }
        #pragma unroll
        for (int j = 0; j < NR_; ++j) {
            b[j].h[0] = *(const v8h*)(bBase[j] + k0);
            b[j].h[1] = *(const v8h*)(bBase[j] + k0 + 16);
        }
        if (k0 + 32 < K) {                       // gfx1250 global_prefetch_b8
            #pragma unroll
            for (int i = 0; i < MR_; ++i) __builtin_prefetch(aBase[i] + k0 + 32, 0, 1);
            #pragma unroll
            for (int j = 0; j < NR_; ++j) __builtin_prefetch(bBase[j] + k0 + 32, 0, 1);
        }
        // (neg_a, A, neg_b, B, c_mod, C, reuse_a, reuse_b)
        #pragma unroll
        for (int i = 0; i < MR_; ++i)
            #pragma unroll
            for (int j = 0; j < NR_; ++j)
                acc[i][j] = __builtin_amdgcn_wmma_f32_16x16x32_f16(
                    false, a[i].v, false, b[j].v, (short)0, acc[i][j],
                    false, false);
    }

    // Epilogue: bias + optional ReLU, f16 store (hidden) or masked f32 store.
    #pragma unroll
    for (int j = 0; j < NR_; ++j) {
        const int n = n0 + j * 16 + lane;
        const float bv = (n < nvalid) ? bias[n] : 0.0f;
        if (!final_f32) {
            _Float16* O = (_Float16*)out;
            #pragma unroll
            for (int i = 0; i < MR_; ++i) {
                #pragma unroll
                for (int r = 0; r < 8; ++r) {
                    const int m = m0 + i * 16 + r + hi * 8;
                    float v = acc[i][j][r] + bv;
                    v = v > 0.0f ? v : 0.0f;     // ReLU
                    O[(size_t)m * ldout + n] = (_Float16)v;
                }
            }
        } else if (n < nvalid) {                 // mask padded columns (117)
            float* O = (float*)out;
            #pragma unroll
            for (int i = 0; i < MR_; ++i) {
                #pragma unroll
                for (int r = 0; r < 8; ++r) {
                    const int m = m0 + i * 16 + r + hi * 8;
                    O[(size_t)m * ldout + n] = acc[i][j][r] + bv;
                }
            }
        }
    }
}

// ---------------------------------------------------------------------------
// Launch
// ---------------------------------------------------------------------------
extern "C" void kernel_launch(void* const* d_in, const int* in_sizes, int n_in,
                              void* d_out, int out_size, void* d_ws, size_t ws_size,
                              hipStream_t stream) {
    const float* feats  = (const float*)d_in[0];
    const int*   hboxes = (const int*)d_in[1];
    const int*   oboxes = (const int*)d_in[2];
    const float* W1     = (const float*)d_in[3];
    const float* b1     = (const float*)d_in[4];
    const float* W2     = (const float*)d_in[5];
    const float* b2     = (const float*)d_in[6];
    const float* W3     = (const float*)d_in[7];
    const float* b3     = (const float*)d_in[8];

    // Workspace layout (all 16B aligned)
    char* ws = (char*)d_ws;
    size_t off = 0;
    float*    roi = (float*)(ws + off);    off += (size_t)B_ * NBOX * D_ * sizeof(float);   // 1.18 MB
    _Float16* A1  = (_Float16*)(ws + off); off += (size_t)M_ * K1_ * sizeof(_Float16);      // 6.29 MB
    _Float16* Wt1 = (_Float16*)(ws + off); off += (size_t)N1_ * K1_ * sizeof(_Float16);     // 1.57 MB
    _Float16* H1  = (_Float16*)(ws + off); off += (size_t)M_ * N1_ * sizeof(_Float16);      // 2.10 MB
    _Float16* Wt2 = (_Float16*)(ws + off); off += (size_t)N2_ * N1_ * sizeof(_Float16);     // 0.26 MB
    _Float16* H2  = (_Float16*)(ws + off); off += (size_t)M_ * N2_ * sizeof(_Float16);      // 1.05 MB
    _Float16* Wt3 = (_Float16*)(ws + off); off += (size_t)N3P_ * N2_ * sizeof(_Float16);    // 0.07 MB

    // 1) ROI means (memory-bound stage; coalesced over D)
    roi_mean_kernel<<<B_ * NBOX, 256, 0, stream>>>(feats, hboxes, oboxes, roi);

    // 2) Pair matrix in f16
    build_pairs_kernel<<<M_, 256, 0, stream>>>(roi, A1);

    // 3) Weight staging (transpose + f16)
    {
        int tot1 = N1_ * K1_;
        transpose_w_kernel<<<(tot1 + 255) / 256, 256, 0, stream>>>(W1, Wt1, K1_, N1_, N1_);
        int tot2 = N2_ * N1_;
        transpose_w_kernel<<<(tot2 + 255) / 256, 256, 0, stream>>>(W2, Wt2, N1_, N2_, N2_);
        int tot3 = N3P_ * N2_;
        transpose_w_kernel<<<(tot3 + 255) / 256, 256, 0, stream>>>(W3, Wt3, N2_, N3_, N3P_);
    }

    // 4) MLP on WMMA: layer1 (2048x1536x512, ReLU) -> layer2 (x512x256, ReLU)
    //    -> layer3 (x256x128, f32 out masked to 117)
    {
        dim3 blk(32, 1, 1);
        dim3 g1(N1_ / (16 * NR_), M_ / (16 * MR_));      // 16 x 32
        wmma_gemm_kernel<<<g1, blk, 0, stream>>>(A1, Wt1, b1, (void*)H1,
                                                 K1_, N1_, N1_, 0);
        dim3 g2(N2_ / (16 * NR_), M_ / (16 * MR_));      // 8 x 32
        wmma_gemm_kernel<<<g2, blk, 0, stream>>>(H1, Wt2, b2, (void*)H2,
                                                 N1_, N2_, N2_, 0);
        dim3 g3(N3P_ / (16 * NR_), M_ / (16 * MR_));     // 4 x 32
        wmma_gemm_kernel<<<g3, blk, 0, stream>>>(H2, Wt3, b3, d_out,
                                                 N2_, N3_, N3_, 1);
    }
}
